// MultiHeadAtt_35038343201434
// MI455X (gfx1250) — matrix operand order repaired and checked by reading, not compiled, not run on hardware
//
#include <hip/hip_runtime.h>
#include <hip/hip_bf16.h>

// ---------------------------------------------------------------------------
// MI455X (gfx1250) multi-head attention, bf16 WMMA path, f32 accumulation.
// Shapes fixed by the reference: B=2, Ms=Mt=2048, D=1024, h=16, dk=64.
// Software-pipelined K loops: next k-tile fragments load while current tile's
// WMMAs issue; prefetch (locality 3 -> WGP scope) pulls the tile after that.
// ---------------------------------------------------------------------------

typedef __bf16 bf16;
typedef __attribute__((ext_vector_type(16))) __bf16 bf16x16;
typedef __attribute__((ext_vector_type(8)))  __bf16 bf16x8;
typedef __attribute__((ext_vector_type(8)))  float  f32x8;
typedef __attribute__((ext_vector_type(4)))  float  f32x4;

static constexpr int BATCH = 2;
static constexpr int MSEQ  = 2048;   // Ms == Mt
static constexpr int DMOD  = 1024;
static constexpr int HEADS = 16;
static constexpr int DK    = 64;
static constexpr int MTOT  = BATCH * MSEQ;          // 4096 rows
static constexpr size_t ATT_ELEMS = (size_t)HEADS * BATCH * MSEQ * MSEQ;

// ---- fragment helpers (layouts per CDNA5 ISA §7.12.2) ----------

union AFragU { bf16x16 v; bf16x8 h[2]; };

// A: 16x32 bf16 tile of row-major [*, lda] matrix at (row0, k0)
__device__ __forceinline__ bf16x16 load_a_frag(const bf16* __restrict__ A, int lda,
                                               int row0, int k0, int lane) {
  const int r = lane & 15, half = lane >> 4;
  const bf16* base = A + (size_t)(row0 + r) * lda + k0 + half * 8;
  AFragU f;
  f.h[0] = *(const bf16x8*)(base);        // K = k0 + half*8 + [0..7]
  f.h[1] = *(const bf16x8*)(base + 16);   // K = k0 + 16 + half*8 + [0..7]
  return f.v;
}

// Raw fp32 A tile (16 floats/lane); converted to bf16 just before the WMMA so
// the cvt VALU overlaps matrix issue instead of sitting behind the loads.
struct AF32 { f32x4 q[4]; };
__device__ __forceinline__ AF32 load_a_raw_f32(const float* __restrict__ A, int lda,
                                               int row0, int k0, int lane) {
  const int r = lane & 15, half = lane >> 4;
  const float* base = A + (size_t)(row0 + r) * lda + k0 + half * 8;
  AF32 f;
  f.q[0] = *(const f32x4*)(base);
  f.q[1] = *(const f32x4*)(base + 4);
  f.q[2] = *(const f32x4*)(base + 16);
  f.q[3] = *(const f32x4*)(base + 20);
  return f;
}
__device__ __forceinline__ bf16x16 cvt_a(const AF32& f) {
  bf16x16 v;
#pragma unroll
  for (int e = 0; e < 16; ++e) v[e] = (bf16)f.q[e >> 2][e & 3];
  return v;
}

// B: 32x16 tile where B[k][n] = W[n][k], W row-major [N, ldw]. Two 16B runs/lane.
__device__ __forceinline__ bf16x16 load_b_frag(const bf16* __restrict__ W, int ldw,
                                               int col0, int k0, int lane) {
  const int n = lane & 15, half = lane >> 4;
  const bf16* base = W + (size_t)(col0 + n) * ldw + k0 + half * 8;
  AFragU f;
  f.h[0] = *(const bf16x8*)(base);
  f.h[1] = *(const bf16x8*)(base + 16);
  return f.v;
}

__device__ __forceinline__ f32x8 wmma_bf16(bf16x16 a, bf16x16 b, f32x8 c) {
  return __builtin_amdgcn_wmma_f32_16x16x32_bf16(false, a, false, b,
                                                 (short)0, c, false, false);
}

// ---- kernel 0: fp32 -> bf16 pack ------------------------------------------

__global__ void mha_pack_bf16(const float* __restrict__ src, bf16* __restrict__ dst, int n) {
  int i = blockIdx.x * blockDim.x + threadIdx.x;
  const int stride = gridDim.x * blockDim.x;
  for (; i < n; i += stride) dst[i] = (bf16)src[i];
}

// ---- kernel 1: projection GEMM  C[M,N] = A[M,K] @ W[N,K]^T + bias ---------
// M=4096, N=1024, K=1024.  vmode==1: scatter into per-head transposed Vt.

__global__ void mha_proj_gemm(const bf16* __restrict__ A, const bf16* __restrict__ W,
                              const float* __restrict__ bias, bf16* __restrict__ C,
                              int vmode, bf16* __restrict__ Vt) {
  const int lane = threadIdx.x & 31;
  const int w    = blockIdx.x * (blockDim.x >> 5) + (threadIdx.x >> 5);
  const int tm = w >> 4, tn = w & 15;           // 256 x 16 wave tiles
  const int row0 = tm * 16, col0 = tn * 64;

  f32x8 acc[4] = {};
  // pipeline prologue
  bf16x16 a0 = load_a_frag(A, DMOD, row0, 0, lane);
  bf16x16 b0[4];
#pragma unroll
  for (int j = 0; j < 4; ++j) b0[j] = load_b_frag(W, DMOD, col0 + j * 16, 0, lane);

  for (int k0 = 0; k0 < DMOD - 32; k0 += 32) {
    const int kn = k0 + 32;
    __builtin_prefetch(A + (size_t)(row0 + (lane & 15)) * DMOD + kn + 32, 0, 3);
    const bf16x16 a1 = load_a_frag(A, DMOD, row0, kn, lane);
    bf16x16 b1[4];
#pragma unroll
    for (int j = 0; j < 4; ++j) b1[j] = load_b_frag(W, DMOD, col0 + j * 16, kn, lane);
#pragma unroll
    for (int j = 0; j < 4; ++j) acc[j] = wmma_bf16(a0, b0[j], acc[j]);
    a0 = a1;
#pragma unroll
    for (int j = 0; j < 4; ++j) b0[j] = b1[j];
  }
#pragma unroll
  for (int j = 0; j < 4; ++j) acc[j] = wmma_bf16(a0, b0[j], acc[j]);

  const int n = lane & 15, half = lane >> 4;
#pragma unroll
  for (int j = 0; j < 4; ++j) {
    const int c = col0 + j * 16 + n;
    const float bia = bias[c];
    if (vmode == 0) {
#pragma unroll
      for (int v = 0; v < 8; ++v) {
        const int m = row0 + v + half * 8;
        C[(size_t)m * DMOD + c] = (bf16)(acc[j][v] + bia);
      }
    } else {
      const int hd = c >> 6, dl = c & 63;
#pragma unroll
      for (int v = 0; v < 8; ++v) {
        const int m = row0 + v + half * 8;
        const int bb = m >> 11, mt = m & (MSEQ - 1);
        // Vt[(hd*B + bb)][dl][mt]
        Vt[(((size_t)(hd * BATCH + bb)) * DK + dl) * MSEQ + mt] = (bf16)(acc[j][v] + bia);
      }
    }
  }
}

// ---- kernel 2: scores = (Qh @ Kh^T) * 1/sqrt(dk)  -> att fp32 -------------
// per pair p = hd*B + b : M=2048, N=2048, K=64 (fully unrolled, 8 WMMAs).

__global__ void mha_score_gemm(const bf16* __restrict__ Qp, const bf16* __restrict__ Kp,
                               float* __restrict__ att) {
  const int lane = threadIdx.x & 31;
  const int w    = blockIdx.x * (blockDim.x >> 5) + (threadIdx.x >> 5);
  const int p  = w >> 12;                // 32 (head,batch) pairs, 4096 waves each
  const int t  = w & 4095;
  const int tm = t >> 5, tn = t & 31;
  const int row0 = tm * 16, col0 = tn * 64;
  const int hd = p >> 1, b = p & 1;

  const bf16* Ab = Qp + (size_t)b * MSEQ * DMOD + hd * DK;
  const bf16* Wb = Kp + (size_t)b * MSEQ * DMOD + hd * DK;

  // load everything first, then a dense WMMA burst
  const bf16x16 aA = load_a_frag(Ab, DMOD, row0, 0, lane);
  const bf16x16 aB = load_a_frag(Ab, DMOD, row0, 32, lane);
  bf16x16 bA[4], bB[4];
#pragma unroll
  for (int j = 0; j < 4; ++j) {
    bA[j] = load_b_frag(Wb, DMOD, col0 + j * 16, 0, lane);
    bB[j] = load_b_frag(Wb, DMOD, col0 + j * 16, 32, lane);
  }
  f32x8 acc[4] = {};
#pragma unroll
  for (int j = 0; j < 4; ++j) acc[j] = wmma_bf16(aA, bA[j], acc[j]);
#pragma unroll
  for (int j = 0; j < 4; ++j) acc[j] = wmma_bf16(aB, bB[j], acc[j]);

  float* Cb = att + (size_t)p * MSEQ * MSEQ;
  const int n = lane & 15, half = lane >> 4;
  const float scale = 0.125f;            // 1/sqrt(64)
#pragma unroll
  for (int j = 0; j < 4; ++j) {
    const int c = col0 + j * 16 + n;
#pragma unroll
    for (int v = 0; v < 8; ++v)
      Cb[(size_t)(row0 + v + half * 8) * MSEQ + c] = acc[j][v] * scale;
  }
}

// ---- kernel 3: row softmax, in place --------------------------------------
// wave32 shuffle reduction + one 8-entry LDS exchange across the 8 waves.

__device__ __forceinline__ float wave_max(float x) {
#pragma unroll
  for (int off = 16; off > 0; off >>= 1) x = fmaxf(x, __shfl_xor(x, off, 32));
  return x;
}
__device__ __forceinline__ float wave_sum(float x) {
#pragma unroll
  for (int off = 16; off > 0; off >>= 1) x += __shfl_xor(x, off, 32);
  return x;
}

__global__ void mha_softmax(float* __restrict__ att) {
  __shared__ float red[8];
  float* p = att + (size_t)blockIdx.x * MSEQ;
  const int t = threadIdx.x;
  const int lane = t & 31, wid = t >> 5;

  float v[8];
  float m = -3.402823466e+38f;
#pragma unroll
  for (int i = 0; i < 8; ++i) { v[i] = p[t + i * 256]; m = fmaxf(m, v[i]); }
  m = wave_max(m);
  if (lane == 0) red[wid] = m;
  __syncthreads();
#pragma unroll
  for (int i = 0; i < 8; ++i) m = fmaxf(m, red[i]);
  __syncthreads();

  float sum = 0.f;
#pragma unroll
  for (int i = 0; i < 8; ++i) { v[i] = __expf(v[i] - m); sum += v[i]; }
  sum = wave_sum(sum);
  if (lane == 0) red[wid] = sum;
  __syncthreads();
  sum = 0.f;
#pragma unroll
  for (int i = 0; i < 8; ++i) sum += red[i];
  const float inv = 1.0f / sum;
#pragma unroll
  for (int i = 0; i < 8; ++i) p[t + i * 256] = v[i] * inv;
}

// ---- kernel 4: res = att @ Vh  (A fp32, cvt in-register, pipelined) -------
// per pair p: M=2048, N=64, K=2048.  Res stored [B*Ms, D] with col = hd*64+dl.

__global__ void mha_av_gemm(const float* __restrict__ att, const bf16* __restrict__ Vt,
                            bf16* __restrict__ Res) {
  const int lane = threadIdx.x & 31;
  const int w    = blockIdx.x * (blockDim.x >> 5) + (threadIdx.x >> 5);
  const int p  = w >> 7;                 // 32 pairs x 128 row tiles
  const int tm = w & 127;
  const int row0 = tm * 16;
  const int hd = p >> 1, b = p & 1;

  const float* Ab = att + (size_t)p * MSEQ * MSEQ;
  const bf16*  Wb = Vt + (size_t)p * DK * MSEQ;   // [64, 2048] row-major

  f32x8 acc[4] = {};
  AF32    a0 = load_a_raw_f32(Ab, MSEQ, row0, 0, lane);
  bf16x16 b0[4];
#pragma unroll
  for (int j = 0; j < 4; ++j) b0[j] = load_b_frag(Wb, MSEQ, j * 16, 0, lane);

  for (int k0 = 0; k0 < MSEQ - 32; k0 += 32) {
    const int kn = k0 + 32;
    __builtin_prefetch(Ab + (size_t)(row0 + (lane & 15)) * MSEQ + kn + 32, 0, 3);
    const AF32 a1 = load_a_raw_f32(Ab, MSEQ, row0, kn, lane);
    bf16x16 b1[4];
#pragma unroll
    for (int j = 0; j < 4; ++j) b1[j] = load_b_frag(Wb, MSEQ, j * 16, kn, lane);
    const bf16x16 ac = cvt_a(a0);
#pragma unroll
    for (int j = 0; j < 4; ++j) acc[j] = wmma_bf16(ac, b0[j], acc[j]);
    a0 = a1;
#pragma unroll
    for (int j = 0; j < 4; ++j) b0[j] = b1[j];
  }
  {
    const bf16x16 ac = cvt_a(a0);
#pragma unroll
    for (int j = 0; j < 4; ++j) acc[j] = wmma_bf16(ac, b0[j], acc[j]);
  }

  const int n = lane & 15, half = lane >> 4;
#pragma unroll
  for (int j = 0; j < 4; ++j) {
    const int c = hd * DK + j * 16 + n;
#pragma unroll
    for (int v = 0; v < 8; ++v) {
      const int m = b * MSEQ + row0 + v + half * 8;
      Res[(size_t)m * DMOD + c] = (bf16)acc[j][v];
    }
  }
}

// ---- kernel 5: out = [query, res] @ Wp^T + bp -----------------------------
// M=4096, N=1024, K=2048 (A = Qa for k<1024, Res for k>=1024), pipelined.

__device__ __forceinline__ bf16x16 load_a_cat(const bf16* __restrict__ Qa,
                                              const bf16* __restrict__ Res,
                                              int row0, int k, int lane) {
  const bf16* Ab = (k < DMOD) ? Qa : Res;     // uniform scalar select
  return load_a_frag(Ab, DMOD, row0, k & (DMOD - 1), lane);
}

__global__ void mha_out_gemm(const bf16* __restrict__ Qa, const bf16* __restrict__ Res,
                             const bf16* __restrict__ Wp, const float* __restrict__ bp,
                             float* __restrict__ out) {
  const int lane = threadIdx.x & 31;
  const int w    = blockIdx.x * (blockDim.x >> 5) + (threadIdx.x >> 5);
  const int tm = w >> 4, tn = w & 15;
  const int row0 = tm * 16, col0 = tn * 64;

  f32x8 acc[4] = {};
  bf16x16 a0 = load_a_cat(Qa, Res, row0, 0, lane);
  bf16x16 b0[4];
#pragma unroll
  for (int j = 0; j < 4; ++j) b0[j] = load_b_frag(Wp, 2 * DMOD, col0 + j * 16, 0, lane);

  for (int k0 = 0; k0 < 2 * DMOD - 32; k0 += 32) {
    const int kn = k0 + 32;
    const bf16x16 a1 = load_a_cat(Qa, Res, row0, kn, lane);
    bf16x16 b1[4];
#pragma unroll
    for (int j = 0; j < 4; ++j) b1[j] = load_b_frag(Wp, 2 * DMOD, col0 + j * 16, kn, lane);
#pragma unroll
    for (int j = 0; j < 4; ++j) acc[j] = wmma_bf16(a0, b0[j], acc[j]);
    a0 = a1;
#pragma unroll
    for (int j = 0; j < 4; ++j) b0[j] = b1[j];
  }
#pragma unroll
  for (int j = 0; j < 4; ++j) acc[j] = wmma_bf16(a0, b0[j], acc[j]);

  const int n = lane & 15, half = lane >> 4;
#pragma unroll
  for (int j = 0; j < 4; ++j) {
    const int c = col0 + j * 16 + n;
    const float bia = bp[c];
#pragma unroll
    for (int v = 0; v < 8; ++v)
      out[(size_t)(row0 + v + half * 8) * DMOD + c] = acc[j][v] + bia;
  }
}

// ---------------------------------------------------------------------------

extern "C" void kernel_launch(void* const* d_in, const int* in_sizes, int n_in,
                              void* d_out, int out_size, void* d_ws, size_t ws_size,
                              hipStream_t stream) {
  (void)in_sizes; (void)n_in; (void)out_size; (void)ws_size;

  const float* key   = (const float*)d_in[0];
  const float* query = (const float*)d_in[1];
  const float* value = (const float*)d_in[2];
  const float* Wk    = (const float*)d_in[3];
  const float* bk    = (const float*)d_in[4];
  const float* Wq    = (const float*)d_in[5];
  const float* bq    = (const float*)d_in[6];
  const float* Wv    = (const float*)d_in[7];
  const float* bv    = (const float*)d_in[8];
  const float* Wp    = (const float*)d_in[9];
  const float* bp    = (const float*)d_in[10];

  float* att = (float*)d_out;                       // [32, 2048, 2048]
  float* out = att + ATT_ELEMS;                     // [2, 2048, 1024]

  // workspace layout (bytes, 256-aligned by construction)
  char* ws = (char*)d_ws;
  const size_t ACT = (size_t)MTOT * DMOD * sizeof(bf16);   // 8 MiB
  const size_t WSZ = (size_t)DMOD * DMOD * sizeof(bf16);   // 2 MiB
  bf16* Qa  = (bf16*)(ws);                                  // query bf16
  bf16* Ka  = (bf16*)(ws + ACT);                            // key bf16
  bf16* Va  = (bf16*)(ws + 2 * ACT);                        // value bf16
  bf16* Wqb = (bf16*)(ws + 3 * ACT);
  bf16* Wkb = (bf16*)(ws + 3 * ACT + WSZ);
  bf16* Wvb = (bf16*)(ws + 3 * ACT + 2 * WSZ);
  bf16* Wpb = (bf16*)(ws + 3 * ACT + 3 * WSZ);              // [1024, 2048]
  bf16* Qp  = (bf16*)(ws + 3 * ACT + 5 * WSZ);              // projected Q
  bf16* Kp  = (bf16*)(ws + 4 * ACT + 5 * WSZ);              // projected K
  bf16* Vt  = (bf16*)(ws + 5 * ACT + 5 * WSZ);              // [32, 64, 2048]
  bf16* Res = (bf16*)(ws + 6 * ACT + 5 * WSZ);              // attn output bf16

  const int NACT = MTOT * DMOD;        // 4 M elems
  const int NW   = DMOD * DMOD;        // 1 M elems
  mha_pack_bf16<<<1024, 256, 0, stream>>>(query, Qa, NACT);
  mha_pack_bf16<<<1024, 256, 0, stream>>>(key,   Ka, NACT);
  mha_pack_bf16<<<1024, 256, 0, stream>>>(value, Va, NACT);
  mha_pack_bf16<<<1024, 256, 0, stream>>>(Wq, Wqb, NW);
  mha_pack_bf16<<<1024, 256, 0, stream>>>(Wk, Wkb, NW);
  mha_pack_bf16<<<1024, 256, 0, stream>>>(Wv, Wvb, NW);
  mha_pack_bf16<<<1024, 256, 0, stream>>>(Wp, Wpb, 2 * NW);

  // projections: 4096 waves -> 512 blocks of 8 wave32s
  mha_proj_gemm<<<512, 256, 0, stream>>>(Qa, Wqb, bq, Qp, 0, Qp);
  mha_proj_gemm<<<512, 256, 0, stream>>>(Ka, Wkb, bk, Kp, 0, Kp);
  mha_proj_gemm<<<512, 256, 0, stream>>>(Va, Wvb, bv, Vt, 1, Vt);

  // scores: 32 pairs x 4096 waves -> 16384 blocks
  mha_score_gemm<<<16384, 256, 0, stream>>>(Qp, Kp, att);

  // softmax: 65536 rows
  mha_softmax<<<65536, 256, 0, stream>>>(att);

  // attention x V: 4096 waves -> 512 blocks
  mha_av_gemm<<<512, 256, 0, stream>>>(att, Vt, Res);

  // output projection
  mha_out_gemm<<<512, 256, 0, stream>>>(Qa, Res, Wpb, bp, out);
}
